// SocialModel_53111565582699
// MI455X (gfx1250) — matrix operand order repaired and checked by reading, not compiled, not run on hardware
//
#include <hip/hip_runtime.h>

typedef __bf16 bf16_t;
typedef __attribute__((ext_vector_type(16))) __bf16 v16bf;
typedef __attribute__((ext_vector_type(8)))  __bf16 v8bf;
typedef __attribute__((ext_vector_type(8)))  float  v8f;

namespace {
constexpr int T_FR = 20;
constexpr int NP   = 512;            // pedestrians
constexpr int G2C  = 16;             // grid cells
constexpr int HC   = 128;            // rnn size
constexpr int EC   = 64;             // embedding size
constexpr int KSOC = NP * G2C;       // 8192 : K of social GEMM
constexpr int XHW  = 4 * EC + HC;    // 384  : width of [x | h] row
constexpr int GW   = 4 * HC;         // 512  : gates width
constexpr int KSLICES = 32;          // one K-slice per wave
constexpr int KPER    = KSOC / KSLICES;  // 256 -> 8 K-steps of 32
constexpr int OUTW = 5;
}

// pack 16 consecutive-K f32 values into a bf16 ext-vector (VGPR pair order)
__device__ __forceinline__ v16bf cvt16(float4 a, float4 b, float4 c, float4 d) {
  v16bf r;
  r[0]=(__bf16)a.x;  r[1]=(__bf16)a.y;  r[2]=(__bf16)a.z;  r[3]=(__bf16)a.w;
  r[4]=(__bf16)b.x;  r[5]=(__bf16)b.y;  r[6]=(__bf16)b.z;  r[7]=(__bf16)b.w;
  r[8]=(__bf16)c.x;  r[9]=(__bf16)c.y;  r[10]=(__bf16)c.z; r[11]=(__bf16)c.w;
  r[12]=(__bf16)d.x; r[13]=(__bf16)d.y; r[14]=(__bf16)d.z; r[15]=(__bf16)d.w;
  return r;
}

__device__ __forceinline__ v8f wmma_bf16(v16bf a, v16bf b, v8f c) {
  return __builtin_amdgcn_wmma_f32_16x16x32_bf16(false, a, false, b, (short)0, c,
                                                 false, false);
}

__device__ __forceinline__ float sigmoidf_(float x) { return 1.0f / (1.0f + expf(-x)); }

// ---------------------------------------------------------------------------
// init: h, c from inputs; seed h-columns of xh (bf16)
__global__ void k_init(const float* __restrict__ h0, const float* __restrict__ c0,
                       float* __restrict__ hC, float* __restrict__ cC,
                       bf16_t* __restrict__ xh) {
  int i = blockIdx.x * blockDim.x + threadIdx.x;  // < 512*128
  float hv = h0[i];
  hC[i] = hv;
  cC[i] = c0[i];
  int n = i >> 7, hh = i & 127;
  xh[(size_t)n * XHW + 4 * EC + hh] = (__bf16)hv;
}

// ---------------------------------------------------------------------------
// Pt[e, m*16+g] = sum_h W_tensor[e, g*128+h] * hC[m, h]   (bf16 output)
// grid: (16 g, 4 m-chunks), block 128 (4 waves = 4 e-tiles of 16)
__global__ void k_social_P(const float* __restrict__ Wt, const float* __restrict__ hC,
                           bf16_t* __restrict__ Pt) {
  const int g    = blockIdx.x;
  const int wave = threadIdx.x >> 5;
  const int lane = threadIdx.x & 31;
  const int half = lane >> 4;
  const int l    = lane & 15;
  const int e0   = wave * 16;

  // A operand: rows e, K = h (invariant over m-tiles -> hoist)
  const float* arow = Wt + (size_t)(e0 + l) * (G2C * HC) + g * HC;
  v16bf A[4];
#pragma unroll
  for (int s = 0; s < 4; ++s) {
    const float* p0 = arow + s * 32 + half * 16;
    const float* p1 = p0 + 8;
    A[s] = cvt16(*(const float4*)p0, *(const float4*)(p0 + 4),
                 *(const float4*)p1, *(const float4*)(p1 + 4));
  }
  const int mt_end = blockIdx.y * 8 + 8;
  for (int mt = blockIdx.y * 8; mt < mt_end; ++mt) {
    const float* brow = hC + (size_t)(mt * 16 + l) * HC;  // B: N = m, K = h
    v8f acc = {};
#pragma unroll
    for (int s = 0; s < 4; ++s) {
      const float* p = brow + s * 32 + half * 16;
      v16bf B = cvt16(*(const float4*)p, *(const float4*)(p + 4),
                      *(const float4*)(p + 8), *(const float4*)(p + 12));
      acc = wmma_bf16(A[s], B, acc);
    }
    const int col = (mt * 16 + l) * G2C + g;
#pragma unroll
    for (int v = 0; v < 8; ++v)
      Pt[(size_t)(e0 + v + 8 * half) * KSOC + col] = (__bf16)acc[v];
  }
}

// ---------------------------------------------------------------------------
// input embeddings: xh[n, which*64+e] = relu(x·W_in + b_in)  (bf16)
__global__ void k_emb(const float* __restrict__ pos, const float* __restrict__ dirn,
                      const float* __restrict__ vel, const float* __restrict__ W_in,
                      const float* __restrict__ b_in, bf16_t* __restrict__ xh, int t) {
  int idx = blockIdx.x * blockDim.x + threadIdx.x;  // < 512*192
  if (idx >= NP * 3 * EC) return;
  int n = idx / (3 * EC), c = idx % (3 * EC);
  int which = c / EC, e = c % EC;
  const float* src = (which == 0) ? pos : (which == 1 ? dirn : vel);
  size_t base = ((size_t)t * NP + n) * 2;
  float v = src[base] * W_in[e * 2] + src[base + 1] * W_in[e * 2 + 1] + b_in[e];
  v = v > 0.f ? v : 0.f;
  xh[(size_t)n * XHW + c] = (__bf16)v;
}

// ---------------------------------------------------------------------------
// social GEMM (dominant): one wave owns a 16x64 output strip for one K-slice.
// A (grid rows, HBM-streamed) loaded ONCE per K-step, amortized over 4 WMMAs.
// grid: (32 n-tiles, 8), block 128; slice = blockIdx.y*4 + wave  (32 slices)
__global__ void k_social(const float* __restrict__ grids, const bf16_t* __restrict__ Pt,
                         float* __restrict__ part, int t) {
  const int wave  = threadIdx.x >> 5;
  const int lane  = threadIdx.x & 31;
  const int half  = lane >> 4;
  const int l     = lane & 15;
  const int n0    = blockIdx.x * 16;
  const int slice = blockIdx.y * 4 + wave;

  const float* arow = grids + ((size_t)t * NP + n0 + l) * KSOC;  // A: M=n, K contiguous
  const bf16_t* b0 = Pt + (size_t)(l) * KSOC;                    // B tiles: e rows
  const bf16_t* b1 = Pt + (size_t)(16 + l) * KSOC;
  const bf16_t* b2 = Pt + (size_t)(32 + l) * KSOC;
  const bf16_t* b3 = Pt + (size_t)(48 + l) * KSOC;
  const int k0 = slice * KPER;

  v8f acc0 = {}, acc1 = {}, acc2 = {}, acc3 = {};
#pragma unroll 2
  for (int kk = k0; kk < k0 + KPER; kk += 32) {
    const float* p0 = arow + kk + half * 16;
    const float* p1 = p0 + 8;
    v16bf A = cvt16(*(const float4*)p0, *(const float4*)(p0 + 4),
                    *(const float4*)p1, *(const float4*)(p1 + 4));
    const int bo = kk + half * 16;
    v16bf B0 = *(const v16bf*)(b0 + bo);   // 32B contiguous bf16 each
    v16bf B1 = *(const v16bf*)(b1 + bo);
    v16bf B2 = *(const v16bf*)(b2 + bo);
    v16bf B3 = *(const v16bf*)(b3 + bo);
    acc0 = wmma_bf16(A, B0, acc0);
    acc1 = wmma_bf16(A, B1, acc1);
    acc2 = wmma_bf16(A, B2, acc2);
    acc3 = wmma_bf16(A, B3, acc3);
  }
  float* prow = part + (size_t)slice * NP * EC;
#pragma unroll
  for (int v = 0; v < 8; ++v) {
    const size_t rb = (size_t)(n0 + v + 8 * half) * EC + l;
    prow[rb]      = acc0[v];
    prow[rb + 16] = acc1[v];
    prow[rb + 32] = acc2[v];
    prow[rb + 48] = acc3[v];
  }
}

__global__ void k_social_reduce(const float* __restrict__ part,
                                const float* __restrict__ b_tensor,
                                bf16_t* __restrict__ xh) {
  int i = blockIdx.x * blockDim.x + threadIdx.x;  // < 512*64
  int n = i >> 6, e = i & 63;
  float s = b_tensor[e];
#pragma unroll
  for (int sl = 0; sl < KSLICES; ++sl) s += part[((size_t)sl * NP + n) * EC + e];
  s = s > 0.f ? s : 0.f;
  xh[(size_t)n * XHW + 3 * EC + e] = (__bf16)s;
}

// ---------------------------------------------------------------------------
// gates[n, r] = xh[n, :] · [W_ih | W_hh][r, :]   (M=512, N=512, K=384)
// grid: (32 n-tiles, 8), block 128 (4 waves -> 32 r-tiles)
__global__ void k_gates(const bf16_t* __restrict__ xh, const float* __restrict__ W_ih,
                        const float* __restrict__ W_hh, float* __restrict__ gates) {
  const int wave = threadIdx.x >> 5;
  const int lane = threadIdx.x & 31;
  const int half = lane >> 4;
  const int l    = lane & 15;
  const int n0   = blockIdx.x * 16;
  const int r0   = (blockIdx.y * 4 + wave) * 16;

  const bf16_t* arow = xh + (size_t)(n0 + l) * XHW;
  const int r = r0 + l;

  union AU { v16bf v; struct { v8bf lo, hi; } p; };

  v8f acc = {};
#pragma unroll
  for (int s = 0; s < 12; ++s) {
    const int kk = s * 32;
    AU A;
    A.p.lo = *(const v8bf*)(arow + kk + 16 * half);        // 16B bf16 loads
    A.p.hi = *(const v8bf*)(arow + kk + 8 + 16 * half);
    const float* bp = (kk < 4 * EC)
                          ? (W_ih + (size_t)r * (4 * EC) + kk + 16 * half)
                          : (W_hh + (size_t)r * HC + (kk - 4 * EC) + 16 * half);
    v16bf B = cvt16(*(const float4*)bp, *(const float4*)(bp + 4),
                    *(const float4*)(bp + 8), *(const float4*)(bp + 12));
    acc = wmma_bf16(A.v, B, acc);
  }
#pragma unroll
  for (int v = 0; v < 8; ++v)
    gates[(size_t)(n0 + v + 8 * half) * GW + r0 + l] = acc[v];
}

// ---------------------------------------------------------------------------
// LSTM pointwise + output projection. one block per node n, 128 threads.
__global__ void k_lstm(const float* __restrict__ gates, const float* __restrict__ b_ih,
                       const float* __restrict__ b_hh, float* __restrict__ hC,
                       float* __restrict__ cC, bf16_t* __restrict__ xh,
                       const float* __restrict__ W_out, const float* __restrict__ b_out,
                       float* __restrict__ out, int t) {
  __shared__ float hs[HC];
  const int n = blockIdx.x, hh = threadIdx.x;
  const size_t gb = (size_t)n * GW;
  float ig = gates[gb + hh]            + b_ih[hh]            + b_hh[hh];
  float fg = gates[gb + HC + hh]       + b_ih[HC + hh]       + b_hh[HC + hh];
  float gg = gates[gb + 2 * HC + hh]   + b_ih[2 * HC + hh]   + b_hh[2 * HC + hh];
  float og = gates[gb + 3 * HC + hh]   + b_ih[3 * HC + hh]   + b_hh[3 * HC + hh];
  float cn = sigmoidf_(fg) * cC[(size_t)n * HC + hh] + sigmoidf_(ig) * tanhf(gg);
  float hn = sigmoidf_(og) * tanhf(cn);
  cC[(size_t)n * HC + hh] = cn;
  hC[(size_t)n * HC + hh] = hn;
  xh[(size_t)n * XHW + 4 * EC + hh] = (__bf16)hn;
  hs[hh] = hn;
  __syncthreads();
  if (hh < OUTW) {
    float s = b_out[hh];
#pragma unroll 16
    for (int k = 0; k < HC; ++k) s += hs[k] * W_out[hh * HC + k];
    out[((size_t)t * NP + n) * OUTW + hh] = s;
  }
}

// ---------------------------------------------------------------------------
__global__ void k_final(const float* __restrict__ hC, const float* __restrict__ cC,
                        float* __restrict__ out) {
  int i = blockIdx.x * blockDim.x + threadIdx.x;  // < 512*128
  out[(size_t)T_FR * NP * OUTW + i] = hC[i];
  out[(size_t)T_FR * NP * OUTW + NP * HC + i] = cC[i];
}

// ---------------------------------------------------------------------------
extern "C" void kernel_launch(void* const* d_in, const int* in_sizes, int n_in,
                              void* d_out, int out_size, void* d_ws, size_t ws_size,
                              hipStream_t stream) {
  const float* in_pos   = (const float*)d_in[0];
  const float* in_dir   = (const float*)d_in[1];
  const float* in_vel   = (const float*)d_in[2];
  const float* grids    = (const float*)d_in[3];
  const float* h0       = (const float*)d_in[4];
  const float* c0       = (const float*)d_in[5];
  const float* W_in     = (const float*)d_in[6];
  const float* b_in     = (const float*)d_in[7];
  const float* W_tensor = (const float*)d_in[8];
  const float* b_tensor = (const float*)d_in[9];
  const float* W_ih     = (const float*)d_in[10];
  const float* b_ih     = (const float*)d_in[11];
  const float* W_hh     = (const float*)d_in[12];
  const float* b_hh     = (const float*)d_in[13];
  const float* W_out    = (const float*)d_in[14];
  const float* b_out    = (const float*)d_in[15];
  float* out = (float*)d_out;

  char* ws = (char*)d_ws;
  size_t off = 0;
  bf16_t* Pt   = (bf16_t*)(ws + off); off += (size_t)EC * KSOC * sizeof(bf16_t);        // 1 MB
  bf16_t* xh   = (bf16_t*)(ws + off); off += (size_t)NP * XHW * sizeof(bf16_t);         // 384 KB
  float* gates = (float*)(ws + off);  off += (size_t)NP * GW * sizeof(float);           // 1 MB
  float* hC    = (float*)(ws + off);  off += (size_t)NP * HC * sizeof(float);           // 256 KB
  float* cC    = (float*)(ws + off);  off += (size_t)NP * HC * sizeof(float);           // 256 KB
  float* part  = (float*)(ws + off);  off += (size_t)KSLICES * NP * EC * sizeof(float); // 4 MB

  k_init<<<NP * HC / 256, 256, 0, stream>>>(h0, c0, hC, cC, xh);

  for (int t = 0; t < T_FR; ++t) {
    k_social_P<<<dim3(G2C, 4), 128, 0, stream>>>(W_tensor, hC, Pt);
    k_emb<<<(NP * 3 * EC + 255) / 256, 256, 0, stream>>>(in_pos, in_dir, in_vel,
                                                         W_in, b_in, xh, t);
    k_social<<<dim3(NP / 16, KSLICES / 4), 128, 0, stream>>>(grids, Pt, part, t);
    k_social_reduce<<<NP * EC / 256, 256, 0, stream>>>(part, b_tensor, xh);
    k_gates<<<dim3(NP / 16, 8), 128, 0, stream>>>(xh, W_ih, W_hh, gates);
    k_lstm<<<NP, HC, 0, stream>>>(gates, b_ih, b_hh, hC, cC, xh, W_out, b_out, out, t);
  }

  k_final<<<NP * HC / 256, 256, 0, stream>>>(hC, cC, out);
}
